// gclstm_53085795778992
// MI455X (gfx1250) — matrix operand order repaired and checked by reading, not compiled, not run on hardware
//
#include <hip/hip_runtime.h>

// ---------------------------------------------------------------------------
// Types for CDNA5 WMMA
// ---------------------------------------------------------------------------
typedef __attribute__((ext_vector_type(16))) __bf16 bf16x16;
typedef __attribute__((ext_vector_type(8)))  __bf16 bf16x8;
typedef __attribute__((ext_vector_type(2)))  __bf16 bf16x2;
typedef __attribute__((ext_vector_type(8)))  float  f32x8;
typedef __attribute__((ext_vector_type(4)))  float  f32x4;

#define Bdim 16
#define Tdim 128
#define Ndim 256
#define Fdim 128
#define Hdim 128
#define ROWS_TOTAL (Bdim * Ndim)          // 4096 state rows
#define ROW_ELEMS  (ROWS_TOTAL * Hdim)    // 524288 h elements per timestep
#define A_STRIDE   264                    // padded LDS row stride (bf16 units)
#define G_STRIDE   132                    // padded LDS gate-row stride (f32)

// fp32 -> bf16 bits, round-to-nearest-even
__device__ __forceinline__ unsigned short f2bfu(float f) {
    unsigned u = __builtin_bit_cast(unsigned, f);
    u = (u + 0x7FFFu + ((u >> 16) & 1u)) >> 16;
    return (unsigned short)u;
}
// Pack two fp32 into a dword of 2 bf16 — native v_cvt_pk_bf16_f32 if present
#if __has_builtin(__builtin_amdgcn_cvt_pk_bf16_f32)
__device__ __forceinline__ unsigned pack2bf(float lo, float hi) {
    bf16x2 r = __builtin_amdgcn_cvt_pk_bf16_f32(lo, hi);
    return __builtin_bit_cast(unsigned, r);
}
#else
__device__ __forceinline__ unsigned pack2bf(float lo, float hi) {
    return ((unsigned)f2bfu(hi) << 16) | f2bfu(lo);
}
#endif
// Fast activations on the native transcendental pipe (v_exp_f32 / v_rcp_f32)
__device__ __forceinline__ float fast_sigmoid(float z) {
    return __fdividef(1.0f, 1.0f + __expf(-z));
}
__device__ __forceinline__ float fast_tanh(float z) {
    const float e = __expf(2.0f * z);
    return __fdividef(e - 1.0f, e + 1.0f);
}

// ---------------------------------------------------------------------------
// Pack the 8 weight matrices ([128,128] row-major fp32 each) into the exact
// per-lane WMMA B-operand layout (bf16), gate order i,f,o,g:
//   flat index = ((kt*32 + ntile)*32 + lane)*16 + j
//   K = kt_local*32 + 16*(lane>=16) + j ;  N = ntile*16 + (lane&15)
//   kt 0..3 -> Wx (input side), kt 4..7 -> Wm (recurrent side)
// ---------------------------------------------------------------------------
__global__ void gclstm_pack_weights(
    const float* __restrict__ Wix, const float* __restrict__ Wfx,
    const float* __restrict__ Wox, const float* __restrict__ Wcx,
    const float* __restrict__ Wim, const float* __restrict__ Wfm,
    const float* __restrict__ Wom, const float* __restrict__ Wcm,
    unsigned short* __restrict__ Wpack)
{
    int idx  = blockIdx.x * blockDim.x + threadIdx.x;   // 0 .. 131071
    int j    = idx & 15;
    int lane = (idx >> 4) & 31;
    int ntg  = (idx >> 9) & 31;
    int kt   = idx >> 14;                               // 0..7
    int klocal = (kt & 3) * 32 + ((lane >> 4) & 1) * 16 + j;
    int ng     = ntg * 16 + (lane & 15);                // 0..511
    int gate   = ng >> 7;                               // 0=i 1=f 2=o 3=g
    int col    = ng & 127;
    const float* W;
    if (kt < 4) W = (gate == 0) ? Wix : (gate == 1) ? Wfx : (gate == 2) ? Wox : Wcx;
    else        W = (gate == 0) ? Wim : (gate == 1) ? Wfm : (gate == 2) ? Wom : Wcm;
    Wpack[idx] = f2bfu(W[klocal * 128 + col]);
}

// ---------------------------------------------------------------------------
// Persistent GC-LSTM: each block owns 16 state rows for all 128 timesteps.
// 256 threads = 8 wave32; wave w computes a 16x64 half-gate block (gate=w/2).
// h state lives in LDS (bf16, the K=128..255 half of the A tile); c state in
// registers.  The compiler hoists all 32 B-fragments into persistent VGPRs
// (extended >256 VGPR file), so the steady state is ds_load + wmma only.
// ---------------------------------------------------------------------------
__global__ __launch_bounds__(256) void gclstm_persistent(
    const float*          __restrict__ x,
    const unsigned short* __restrict__ Wpack,
    const float* __restrict__ bi, const float* __restrict__ bfg,
    const float* __restrict__ bo, const float* __restrict__ bc,
    float*       __restrict__ out)
{
    __shared__ unsigned short A_lds[16 * A_STRIDE];       // [16][256] bf16: x | h
    __shared__ float          G_lds[4 * 16 * G_STRIDE];   // 4 gates x [16][128] f32

    const int tid  = threadIdx.x;
    const int r0   = blockIdx.x * 16;         // 16 consecutive rows, same b
    const int b    = r0 / Ndim;
    const int n0   = r0 % Ndim;

    const int wave = tid >> 5;                // 0..7
    const int lane = tid & 31;
    const int lhi  = lane >> 4;
    const int l15  = lane & 15;
    const int gate = wave >> 1;               // 0=i 1=f 2=o 3=g
    const int half = wave & 1;                // which 64-col half of the gate

    // Per-wave bias registers
    const float* bias_ptr = (gate == 0) ? bi : (gate == 1) ? bfg
                          : (gate == 2) ? bo : bc;
    float bv[4];
    #pragma unroll
    for (int nt = 0; nt < 4; ++nt) bv[nt] = bias_ptr[half * 64 + nt * 16 + l15];

    // Staging / epilogue slice: row rl, 8-element chunk
    const int rl  = tid >> 4;                 // 0..15
    const int ch8 = (tid & 15) * 8;           // 0,8,...,120

    // c state in registers, h state (LDS) zero-initialized
    float c_reg[8];
    #pragma unroll
    for (int j = 0; j < 8; ++j) c_reg[j] = 0.0f;
    {
        unsigned* hz = (unsigned*)&A_lds[rl * A_STRIDE + 128 + ch8];
        hz[0] = 0u; hz[1] = 0u; hz[2] = 0u; hz[3] = 0u;
    }

    // This thread's x staging source (advances by N*F per timestep)
    const float* xrow = x + (((size_t)b * Tdim) * Ndim + (n0 + rl)) * Fdim + ch8;
    const size_t xstep = (size_t)Ndim * Fdim;

    // A-fragment LDS base for this lane (contiguous K = 16*(lane>=16) + j, matches Wpack)
    const int abase0 = l15 * A_STRIDE + lhi * 16;

    for (int t = 0; t < Tdim; ++t) {
        // ---- stage x_t (fp32 -> bf16) into A_lds cols [0,128) ----
        const float* xp = xrow + (size_t)t * xstep;
        if (t + 1 < Tdim) __builtin_prefetch(xp + xstep, 0, 1);
        f32x4 v0 = ((const f32x4*)xp)[0];
        f32x4 v1 = ((const f32x4*)xp)[1];
        unsigned* dstx = (unsigned*)&A_lds[rl * A_STRIDE + ch8];
        dstx[0] = pack2bf(v0.x, v0.y);
        dstx[1] = pack2bf(v0.z, v0.w);
        dstx[2] = pack2bf(v1.x, v1.y);
        dstx[3] = pack2bf(v1.z, v1.w);
        __syncthreads();

        // ---- GEMM: K=256 (x-half with Wx, h-half with Wm) ----
        f32x8 acc[4];
        #pragma unroll
        for (int i = 0; i < 4; ++i) acc[i] = (f32x8)(0.0f);

        union Afrag { bf16x16 v; bf16x8 h[2]; };
        Afrag a_cur, a_nxt;
        a_cur.h[0] = *(const bf16x8*)&A_lds[abase0];
        a_cur.h[1] = *(const bf16x8*)&A_lds[abase0 + 8];

        #pragma unroll
        for (int kt = 0; kt < 8; ++kt) {
            // Prefetch next A fragment so its ds_load pair overlaps this
            // stage's 4 WMMAs on the XDL pipe.
            if (kt < 7) {
                const int nb = abase0 + (kt + 1) * 32;
                a_nxt.h[0] = *(const bf16x8*)&A_lds[nb];
                a_nxt.h[1] = *(const bf16x8*)&A_lds[nb + 8];
            }
            #pragma unroll
            for (int nt = 0; nt < 4; ++nt) {
                const int ntg = wave * 4 + nt;
                const bf16x16 bm =
                    *(const bf16x16*)(Wpack + (((kt * 32 + ntg) * 32 + lane) << 4));
                acc[nt] = __builtin_amdgcn_wmma_f32_16x16x32_bf16(
                    false, a_cur.v, false, bm, (short)0, acc[nt], false, false);
            }
            a_cur = a_nxt;
        }

        // ---- bias + activation -> G_lds ----
        #pragma unroll
        for (int nt = 0; nt < 4; ++nt) {
            const int colw = half * 64 + nt * 16 + l15;   // col within gate
            #pragma unroll
            for (int r = 0; r < 8; ++r) {
                const float z = acc[nt][r] + bv[nt];
                const float av = (gate == 3) ? fast_tanh(z) : fast_sigmoid(z);
                G_lds[gate * (16 * G_STRIDE) + (r + 8 * lhi) * G_STRIDE + colw] = av;
            }
        }
        __syncthreads();

        // ---- elementwise c/h update; h -> out (fp32, NT) and back into A ----
        {
            float hn[8];
            const int gb0 = rl * G_STRIDE + ch8;
            #pragma unroll
            for (int j = 0; j < 8; ++j) {
                const float iv = G_lds[gb0 + j];
                const float fv = G_lds[16 * G_STRIDE + gb0 + j];
                const float ov = G_lds[2 * 16 * G_STRIDE + gb0 + j];
                const float gv = G_lds[3 * 16 * G_STRIDE + gb0 + j];
                const float cn = fv * c_reg[j] + iv * gv;
                c_reg[j] = cn;
                hn[j] = ov * fast_tanh(cn);
            }
            // out layout == [T,B,N,H] (the reference's raw reshape).
            // Non-temporal: pure write stream, keep it out of L2's way.
            f32x4* op = (f32x4*)(out + (size_t)t * ROW_ELEMS
                                     + (size_t)(r0 + rl) * Hdim + ch8);
            f32x4 o0 = { hn[0], hn[1], hn[2], hn[3] };
            f32x4 o1 = { hn[4], hn[5], hn[6], hn[7] };
            __builtin_nontemporal_store(o0, op);
            __builtin_nontemporal_store(o1, op + 1);
            // h (bf16) becomes next step's A operand, cols [128,256)
            unsigned* hdst = (unsigned*)&A_lds[rl * A_STRIDE + 128 + ch8];
            hdst[0] = pack2bf(hn[0], hn[1]);
            hdst[1] = pack2bf(hn[2], hn[3]);
            hdst[2] = pack2bf(hn[4], hn[5]);
            hdst[3] = pack2bf(hn[6], hn[7]);
        }
        // next iteration's leading __syncthreads orders h writes before GEMM reads
    }
}

// ---------------------------------------------------------------------------
// Host launcher
// ---------------------------------------------------------------------------
extern "C" void kernel_launch(void* const* d_in, const int* in_sizes, int n_in,
                              void* d_out, int out_size, void* d_ws, size_t ws_size,
                              hipStream_t stream)
{
    (void)in_sizes; (void)n_in; (void)out_size; (void)ws_size;
    const float* x   = (const float*)d_in[0];
    const float* Wix = (const float*)d_in[1];
    const float* Wim = (const float*)d_in[2];
    const float* Wfx = (const float*)d_in[3];
    const float* Wfm = (const float*)d_in[4];
    const float* Wcx = (const float*)d_in[5];
    const float* Wcm = (const float*)d_in[6];
    const float* Wox = (const float*)d_in[7];
    const float* Wom = (const float*)d_in[8];
    const float* bi  = (const float*)d_in[9];
    const float* bfg = (const float*)d_in[10];
    const float* bc  = (const float*)d_in[11];
    const float* bo  = (const float*)d_in[12];

    unsigned short* Wpack = (unsigned short*)d_ws;   // 256 KB packed bf16 weights

    gclstm_pack_weights<<<512, 256, 0, stream>>>(Wix, Wfx, Wox, Wcx,
                                                 Wim, Wfm, Wom, Wcm, Wpack);
    gclstm_persistent<<<ROWS_TOTAL / 16, 256, 0, stream>>>(
        x, Wpack, bi, bfg, bo, bc, (float*)d_out);
}
